// MyAttention_76579266888453
// MI455X (gfx1250) — matrix operand order repaired and checked by reading, not compiled, not run on hardware
//
#include <hip/hip_runtime.h>

// ---------------------------------------------------------------------------
// MI455X (gfx1250): 4x 2048x1024x1024 GEMMs on the bf16 WMMA path
// (v_wmma_f32_16x16x32_bf16, f32 accumulate) + tiny per-batch softmax.
// Round 2: software-pipelined (double-buffered) fragment loads so WMMAs
// overlap in-flight global_load_b128s instead of s_wait_loadcnt 0x0 stalls.
// ---------------------------------------------------------------------------

typedef __attribute__((ext_vector_type(16))) __bf16       v16bf;
typedef __attribute__((ext_vector_type(8)))  float        v8f;
typedef __attribute__((ext_vector_type(4)))  unsigned int v4u;

union BFrag { v4u u[2]; v16bf v; };

namespace {
constexpr int   kB     = 2048;   // batch
constexpr int   kC     = 1024;   // channels
constexpr int   kH     = 16;     // heads
constexpr int   kHD    = 64;     // head dim
constexpr int   kNP    = 64;     // prompts
constexpr int   kL     = 65;     // kv length (1 + prompts)
constexpr int   kKT    = 32;     // number of K-chunks of 32 (1024/32)
constexpr float kScale = 0.125f; // 64^-0.5
}

// ---------------------------------------------------------------------------
// Weight packing: W[k][n] (row-major [in=K, out=N]) -> per-(ntile,ktile,lane)
// contiguous 16-bf16 B fragments so the GEMM hot loop loads B with two b128s.
// B k order used here (contiguous, same as A): lanes 0-15 hold K=0..15 (elem j -> K=j),
// lanes 16-31 hold K=16..31; N = lane & 15.
// ---------------------------------------------------------------------------
__global__ void pack_w_kernel(const float* __restrict__ Wq,
                              const float* __restrict__ Wk,
                              const float* __restrict__ Wv,
                              const float* __restrict__ Wp,
                              __bf16* __restrict__ packed) {
  int t    = blockIdx.x * blockDim.x + threadIdx.x;  // (w,nt,kt,lane)
  int lane = t & 31;
  int kt   = (t >> 5) & 31;
  int ntg  = t >> 10;                                 // 0..255 (4 weights * 64 ntiles)
  if (ntg >= 256) return;
  int w = ntg >> 6;
  const float* W = (w == 0) ? Wq : (w == 1) ? Wk : (w == 2) ? Wv : Wp;
  int n     = ((ntg & 63) << 4) | (lane & 15);
  int kbase = kt * 32 + ((lane >= 16) ? 16 : 0);
  __bf16* dst = packed + (((size_t)(ntg * 32 + kt)) * 32 + lane) * 16;
#pragma unroll
  for (int j = 0; j < 16; ++j) {
    dst[j] = (__bf16)W[(size_t)(kbase + j) * kC + n];
  }
}

__global__ void f32_to_bf16_kernel(const float* __restrict__ src,
                                   __bf16* __restrict__ dst, int n) {
  int i = blockIdx.x * blockDim.x + threadIdx.x;
  if (i < n) dst[i] = (__bf16)src[i];
}

// ---------------------------------------------------------------------------
// One wave computes a 16x64 tile. Double-buffered in registers: loads for
// K-chunk kt+1 are issued before the 4 WMMAs of chunk kt, so the compiler
// can wait on partial loadcnt (10 in flight) instead of draining to zero.
// A k order used here (contiguous, matches packed B): lanes 0-15 = rows 0..15 with K 0-15;
// lanes 16-31 = rows 0..15 with K 16-31. Row-major bf16 A -> two b128s.
// ---------------------------------------------------------------------------
__device__ __forceinline__ void gemm_tile16x64(const __bf16* __restrict__ A,
                                               const __bf16* __restrict__ Wpk,
                                               int mTile, int n64, v8f c[4]) {
  const int lane  = threadIdx.x & 31;
  const int lrow  = lane & 15;
  const int khalf = lane >> 4;
  const __bf16* arow  = A + (size_t)(mTile * 16 + lrow) * kC + khalf * 16;
  // fragment (f, kt) at bbase + (f*kKT + kt)*512 elements
  const __bf16* bbase = Wpk + ((size_t)(n64 * 4) * kKT * 32 + lane) * 16;

  BFrag a[2];
  BFrag b[2][4];

  // Prologue: stage 0 (kt = 0)
  a[0].u[0] = *(const v4u*)(arow);
  a[0].u[1] = *(const v4u*)(arow + 8);
#pragma unroll
  for (int f = 0; f < 4; ++f) {
    const __bf16* bp = bbase + (size_t)f * kKT * 512;
    b[0][f].u[0] = *(const v4u*)(bp);
    b[0][f].u[1] = *(const v4u*)(bp + 8);
  }

#pragma unroll 2
  for (int kt = 0; kt < kKT; ++kt) {
    const int cur = kt & 1;
    const int nxt = cur ^ 1;
    if (kt + 1 < kKT) {
      // Issue next-stage loads BEFORE consuming current fragments.
      a[nxt].u[0] = *(const v4u*)(arow + (kt + 1) * 32);
      a[nxt].u[1] = *(const v4u*)(arow + (kt + 1) * 32 + 8);
#pragma unroll
      for (int f = 0; f < 4; ++f) {
        const __bf16* bp = bbase + ((size_t)f * kKT + (kt + 1)) * 512;
        b[nxt][f].u[0] = *(const v4u*)(bp);
        b[nxt][f].u[1] = *(const v4u*)(bp + 8);
      }
    }
#pragma unroll
    for (int f = 0; f < 4; ++f) {
      c[f] = __builtin_amdgcn_wmma_f32_16x16x32_bf16(
          false, a[cur].v, false, b[cur][f].v, (short)0, c[f], false, false);
    }
  }
}

// Generic GEMM: Out has one [M,1024] f32 slab per packed weight section.
__global__ void gemm_kernel(const __bf16* __restrict__ A,
                            const __bf16* __restrict__ Wpk,
                            float* __restrict__ Out,
                            int Mtiles, int numN64, long long slabStride) {
  const int waveId = blockIdx.x * (blockDim.x >> 5) + (threadIdx.x >> 5);
  if (waveId >= Mtiles * numN64) return;  // wave-uniform: EXEC stays all-1s
  const int mTile = waveId / numN64;
  const int n64   = waveId % numN64;

  v8f c[4];
#pragma unroll
  for (int f = 0; f < 4; ++f) c[f] = {};

  gemm_tile16x64(A, Wpk, mTile, n64, c);

  const int lane  = threadIdx.x & 31;
  const int lrow  = lane & 15;
  const int khalf = lane >> 4;
  const int rowBase = mTile * 16 + khalf * 8;
#pragma unroll
  for (int f = 0; f < 4; ++f) {
    const int ntg = n64 * 4 + f;
    const int w   = ntg >> 6;
    const int col = ((ntg & 63) << 4) | lrow;
    float* o = Out + (long long)w * slabStride + col;
#pragma unroll
    for (int r = 0; r < 8; ++r) {
      o[(long long)(rowBase + r) * kC] = c[f][r];
    }
  }
}

// Projection GEMM with fused bias + residual epilogue -> d_out (x_cls).
__global__ void gemm_proj_kernel(const __bf16* __restrict__ A,
                                 const __bf16* __restrict__ Wpk,
                                 const float* __restrict__ bias,
                                 const float* __restrict__ resid,
                                 float* __restrict__ Out, int Mtiles) {
  const int numN64 = 16;  // N = 1024
  const int waveId = blockIdx.x * (blockDim.x >> 5) + (threadIdx.x >> 5);
  if (waveId >= Mtiles * numN64) return;
  const int mTile = waveId / numN64;
  const int n64   = waveId % numN64;

  v8f c[4];
#pragma unroll
  for (int f = 0; f < 4; ++f) c[f] = {};

  gemm_tile16x64(A, Wpk, mTile, n64, c);

  const int lane  = threadIdx.x & 31;
  const int lrow  = lane & 15;
  const int khalf = lane >> 4;
  const int rowBase = mTile * 16 + khalf * 8;
#pragma unroll
  for (int f = 0; f < 4; ++f) {
    const int col = ((n64 * 4 + f) << 4) | lrow;
    const float bcol = bias[col];
#pragma unroll
    for (int r = 0; r < 8; ++r) {
      const long long idx = (long long)(rowBase + r) * kC + col;
      Out[idx] = c[f][r] + bcol + resid[idx];
    }
  }
}

// ---------------------------------------------------------------------------
// Attention: one block per batch. q (pre-scaled), kx, vx staged in LDS;
// shared Kp/Vp (256 KB each in f32) stay L2-resident across all 2048 blocks.
// ---------------------------------------------------------------------------
__global__ void attn_kernel(const float* __restrict__ Q,
                            const float* __restrict__ Kx,
                            const float* __restrict__ Vx,
                            const float* __restrict__ Kp,
                            const float* __restrict__ Vp,
                            float* __restrict__ rawOut,
                            __bf16* __restrict__ XclsBf) {
  __shared__ float qS[kC];
  __shared__ float kxS[kC];
  __shared__ float vxS[kC];
  __shared__ float rawS[kH * kL];
  __shared__ float attnS[kH * kL];

  const int b   = blockIdx.x;
  const int tid = threadIdx.x;

  for (int i = tid; i < kC; i += 256) {
    qS[i]  = Q[(size_t)b * kC + i] * kScale;  // SCALE folded into q (as reference)
    kxS[i] = Kx[(size_t)b * kC + i];
    vxS[i] = Vx[(size_t)b * kC + i];
  }
  __syncthreads();

  // raw_attn[h][l] = q_h . k_l   (l==0 -> own x row; l>=1 -> shared prompts)
  for (int idx = tid; idx < kH * kL; idx += 256) {
    const int h = idx / kL;
    const int l = idx % kL;
    const int base = h * kHD;
    float acc = 0.f;
    if (l == 0) {
#pragma unroll 8
      for (int d = 0; d < kHD; ++d) acc += qS[base + d] * kxS[base + d];
    } else {
      const float* kr = Kp + (size_t)(l - 1) * kC;
#pragma unroll 8
      for (int d = 0; d < kHD; ++d) acc += qS[base + d] * kr[base + d];
    }
    rawS[idx] = acc;
    rawOut[(size_t)b * (kH * kL) + idx] = acc;  // [B,H,1,L]
  }
  __syncthreads();

  // softmax(raw) * SCALE, one thread per head (tiny: 65 elems)
  if (tid < kH) {
    const int base = tid * kL;
    float m = rawS[base];
    for (int l = 1; l < kL; ++l) m = fmaxf(m, rawS[base + l]);
    float s = 0.f;
    for (int l = 0; l < kL; ++l) s += expf(rawS[base + l] - m);
    const float inv = kScale / s;
    for (int l = 0; l < kL; ++l) attnS[base + l] = expf(rawS[base + l] - m) * inv;
  }
  __syncthreads();

  // x_cls[c] = attn . V column c  (coalesced over c across threads)
  for (int c = tid; c < kC; c += 256) {
    const int h = c >> 6;
    float acc = attnS[h * kL] * vxS[c];
#pragma unroll 8
    for (int r = 0; r < kNP; ++r) acc += attnS[h * kL + 1 + r] * Vp[(size_t)r * kC + c];
    XclsBf[(size_t)b * kC + c] = (__bf16)acc;
  }
}

// ---------------------------------------------------------------------------
extern "C" void kernel_launch(void* const* d_in, const int* in_sizes, int n_in,
                              void* d_out, int out_size, void* d_ws, size_t ws_size,
                              hipStream_t stream) {
  (void)in_sizes; (void)n_in; (void)out_size; (void)ws_size;

  const float* x       = (const float*)d_in[0];  // [2048,1,1024]
  const float* prompts = (const float*)d_in[1];  // [1,64,1024]
  const float* Wq      = (const float*)d_in[2];
  const float* Wk      = (const float*)d_in[3];
  const float* Wv      = (const float*)d_in[4];
  const float* Wp      = (const float*)d_in[5];
  const float* bp      = (const float*)d_in[6];

  float* out_cls = (float*)d_out;                       // [2048,1024]
  float* out_raw = out_cls + (size_t)kB * kC;           // [2048,16,65]

  // Workspace layout (~41 MB)
  char* ws = (char*)d_ws;
  size_t off = 0;
  __bf16* packedW = (__bf16*)(ws + off); off += (size_t)4 * kC * kC * 2;   // 8 MB
  __bf16* Xbf     = (__bf16*)(ws + off); off += (size_t)kB * kC * 2;       // 4 MB
  __bf16* Pbf     = (__bf16*)(ws + off); off += (size_t)kNP * kC * 2;      // 128 KB
  float*  QKV     = (float*)(ws + off);  off += (size_t)3 * kB * kC * 4;   // 24 MB
  float*  KpVp    = (float*)(ws + off);  off += (size_t)2 * kNP * kC * 4;  // 512 KB
  __bf16* XclsBf  = (__bf16*)(ws + off); off += (size_t)kB * kC * 2;       // 4 MB

  // 1) Pack 4 weights into WMMA B-fragment layout (bf16).
  pack_w_kernel<<<1024, 256, 0, stream>>>(Wq, Wk, Wv, Wp, packedW);
  // 2) bf16 copies of x and prompts (A operands).
  f32_to_bf16_kernel<<<(kB * kC) / 256, 256, 0, stream>>>(x, Xbf, kB * kC);
  f32_to_bf16_kernel<<<(kNP * kC) / 256, 256, 0, stream>>>(prompts, Pbf, kNP * kC);
  // 3) Shared Kp/Vp = prompts @ [Wk|Wv]   (M=64, N=2048)
  gemm_kernel<<<16, 256, 0, stream>>>(Pbf, packedW + (size_t)kC * kC, KpVp,
                                      kNP / 16, 32, (long long)kNP * kC);
  // 4) Q/Kx/Vx = X @ [Wq|Wk|Wv]          (M=2048, N=3072)
  gemm_kernel<<<768, 256, 0, stream>>>(Xbf, packedW, QKV,
                                       kB / 16, 48, (long long)kB * kC);
  // 5) Attention + softmax -> raw_attn (d_out) and bf16 x_cls (ws)
  attn_kernel<<<kB, 256, 0, stream>>>(QKV, QKV + (size_t)kB * kC,
                                      QKV + (size_t)2 * kB * kC,
                                      KpVp, KpVp + (size_t)kNP * kC,
                                      out_raw, XclsBf);
  // 6) out = x_cls @ Wp + bp + x  -> d_out
  gemm_proj_kernel<<<256, 256, 0, stream>>>(XclsBf, packedW + (size_t)3 * kC * kC,
                                            bp, x, out_cls, kB / 16);
}
